// GCN_20985210208434
// MI455X (gfx1250) — matrix operand order repaired and checked by reading, not compiled, not run on hardware
//
#include <hip/hip_runtime.h>
#include <stdint.h>
#include <stddef.h>

#define D 128

typedef _Float16 half16 __attribute__((ext_vector_type(16)));
typedef _Float16 half8  __attribute__((ext_vector_type(8)));
typedef float    float8 __attribute__((ext_vector_type(8)));

union H16 { half16 v; half8 h[2]; };

// ---------------- utility kernels ----------------

__global__ __launch_bounds__(256) void fill_f32(float* __restrict__ p, float val, int n) {
    int i = blockIdx.x * 256 + threadIdx.x;
    if (i < n) p[i] = val;
}

__global__ __launch_bounds__(256) void deg_accum(float* __restrict__ deg,
                                                 const long long* __restrict__ dstIdx, int E) {
    int e = blockIdx.x * 256 + threadIdx.x;
    if (e < E) atomicAdd(&deg[(int)dstIdx[e]], 1.0f);
}

__global__ __launch_bounds__(256) void deg_finalize(const float* __restrict__ deg,
                                                    float* __restrict__ dis,
                                                    float* __restrict__ dinv, int n) {
    int i = blockIdx.x * 256 + threadIdx.x;
    if (i < n) {
        float d = deg[i];
        dis[i]  = rsqrtf(d);
        dinv[i] = 1.0f / d;
    }
}

// Wt[n*128 + k] = (f16) W[k*128 + n]  (transposed so B-fragment reads are contiguous)
__global__ __launch_bounds__(256) void convert_w_t(const float* __restrict__ W,
                                                   _Float16* __restrict__ Wt) {
    int i = blockIdx.x * 256 + threadIdx.x;   // 16384 threads
    int k = i >> 7, n = i & 127;
    Wt[n * D + k] = (_Float16)W[k * D + n];
}

__global__ __launch_bounds__(256) void convert_f32_to_f16(const float* __restrict__ x,
                                                          _Float16* __restrict__ xh, int n) {
    int i = blockIdx.x * 256 + threadIdx.x;
    if (i < n) xh[i] = (_Float16)x[i];
}

// ---------------- WMMA GEMM: C[N,128] = A[N,128](f16) x W[128,128](f16, transposed) ----------------
// Block = 256 threads = 8 waves. W (32KB f16, transposed) is staged into LDS once per
// block, then each wave computes a 16-row x 128-col stripe: 8 f32 C fragments, K loop in
// 4 steps of 32. Per K-step: 1 A fragment (global) + all 8 B fragments (ds_load_b128),
// then 8 back-to-back v_wmma_f32_16x16x32_f16 (independent accumulators, no D->A/B hazard).
__global__ __launch_bounds__(256) void gemm_wmma(const _Float16* __restrict__ A,
                                                 const _Float16* __restrict__ Wt,
                                                 float* __restrict__ C,
                                                 int nRowTiles) {
    __shared__ _Float16 lw[D * D];   // 32 KB of the WGP's 320 KB LDS

    // cooperative 16B-granule copy of Wt into LDS: 2048 uint4, 256 threads x 8
    {
        const uint4* src = (const uint4*)Wt;
        uint4*       dst = (uint4*)lw;
        int t = threadIdx.x;
#pragma unroll
        for (int i = 0; i < 8; ++i) dst[t + 256 * i] = src[t + 256 * i];
    }
    __syncthreads();

    const int wave = threadIdx.x >> 5;
    const int lane = threadIdx.x & 31;
    const int tile = blockIdx.x * (blockDim.x >> 5) + wave;
    if (tile >= nRowTiles) return;

    const int colLane = lane & 15;
    const int hiHalf  = lane >> 4;            // 0 for lanes 0-15, 1 for lanes 16-31

    // A fragment lane mapping: row M = lane%16; halves 0..7 -> K = kbaseA..+7,
    // halves 8..15 -> K = kbaseA+8..+15, with kbaseA = ks + 16*hiHalf.
    const _Float16* arow = A + (size_t)((tile << 4) + colLane) * D;

    float8 acc[8];
#pragma unroll
    for (int j = 0; j < 8; ++j) acc[j] = (float8)0.0f;

#pragma unroll
    for (int ks = 0; ks < D; ks += 32) {
        const int kbaseA = ks + (hiHalf << 4);
        H16 ua;
        ua.h[0] = *(const half8*)(arow + kbaseA);
        ua.h[1] = *(const half8*)(arow + kbaseA + 8);

        // B fragment lane mapping: col N = lane%16; half h -> K = ks + 16*hiHalf + h
        const int kbaseB = ks + (hiHalf << 4);
        H16 ub[8];
#pragma unroll
        for (int j = 0; j < 8; ++j) {
            const half8* wp = (const half8*)(lw + ((j << 4) + colLane) * D + kbaseB);
            ub[j].h[0] = wp[0];
            ub[j].h[1] = wp[1];
        }
#pragma unroll
        for (int j = 0; j < 8; ++j) {
            acc[j] = __builtin_amdgcn_wmma_f32_16x16x32_f16(
                false, ua.v, false, ub[j].v, (short)0, acc[j], false, false);
        }
    }

    // C/D fragment: VGPR r -> row M = r + 8*hiHalf, col N = lane%16
    const int rowBase = (tile << 4) + (hiHalf << 3);
#pragma unroll
    for (int j = 0; j < 8; ++j) {
#pragma unroll
        for (int r = 0; r < 8; ++r) {
            C[(size_t)(rowBase + r) * D + (j << 4) + colLane] = acc[j][r];
        }
    }
}

// ---------------- edge scatter: agg[dst] += h[src] * dis[src]*dis[dst] ----------------
// One wave per edge; each lane handles 4 contiguous floats (16B load + 4 atomic f32 adds).
__global__ __launch_bounds__(256) void scatter_edges(const float* __restrict__ h,
                                                     const long long* __restrict__ srcI,
                                                     const long long* __restrict__ dstI,
                                                     const float* __restrict__ dis,
                                                     float* __restrict__ agg, int E) {
    long long gid = (long long)blockIdx.x * 256 + threadIdx.x;
    int e    = (int)(gid >> 5);
    int lane = (int)(gid & 31);
    if (e >= E) return;
    int s = (int)srcI[e];
    int d = (int)dstI[e];
    float norm = dis[s] * dis[d];
    int c0 = lane << 2;
    const float4 v = *(const float4*)(h + (size_t)s * D + c0);
    float* ap = agg + (size_t)d * D + c0;
    atomicAdd(ap + 0, v.x * norm);
    atomicAdd(ap + 1, v.y * norm);
    atomicAdd(ap + 2, v.z * norm);
    atomicAdd(ap + 3, v.w * norm);
}

// ---------------- epilogues ----------------

// out_f16 = (f16) relu(agg + h*dinv[row] + b[col])   (feeds next GEMM)
__global__ __launch_bounds__(256) void combine_relu_f16(const float* __restrict__ agg,
                                                        const float* __restrict__ h,
                                                        const float* __restrict__ dinv,
                                                        const float* __restrict__ b,
                                                        _Float16* __restrict__ outh, int n) {
    int i = blockIdx.x * 256 + threadIdx.x;
    if (i < n) {
        int row = i >> 7, col = i & 127;
        float v = agg[i] + h[i] * dinv[row] + b[col];
        outh[i] = (_Float16)fmaxf(v, 0.0f);
    }
}

// out = relu(agg + h*dinv[row] + b2[col] + res + bfc[col])
__global__ __launch_bounds__(256) void combine_final(const float* __restrict__ agg,
                                                     const float* __restrict__ h,
                                                     const float* __restrict__ dinv,
                                                     const float* __restrict__ b2,
                                                     const float* __restrict__ res,
                                                     const float* __restrict__ bfc,
                                                     float* __restrict__ out, int n) {
    int i = blockIdx.x * 256 + threadIdx.x;
    if (i < n) {
        int row = i >> 7, col = i & 127;
        float v = agg[i] + h[i] * dinv[row] + b2[col] + res[i] + bfc[col];
        out[i] = fmaxf(v, 0.0f);
    }
}

// ---------------- host-side launch ----------------

static inline int gb(long long n) { return (int)((n + 255) / 256); }

extern "C" void kernel_launch(void* const* d_in, const int* in_sizes, int n_in,
                              void* d_out, int out_size, void* d_ws, size_t ws_size,
                              hipStream_t stream) {
    const float*     x    = (const float*)d_in[0];
    const long long* ei   = (const long long*)d_in[1];   // int64 (2, E)
    const float*     W1   = (const float*)d_in[2];
    const float*     b1   = (const float*)d_in[3];
    const float*     W2   = (const float*)d_in[4];
    const float*     b2   = (const float*)d_in[5];
    const float*     Wfc  = (const float*)d_in[6];
    const float*     bfc  = (const float*)d_in[7];
    float*           out  = (float*)d_out;

    const int N = in_sizes[0] / D;   // 50000
    const int E = in_sizes[1] / 2;   // 800000
    const long long* srcI = ei;
    const long long* dstI = ei + E;

    // workspace carve-out (256B aligned slices)
    char* w = (char*)d_ws;
    auto alloc = [&](size_t bytes) -> void* {
        void* p = (void*)w;
        w += (bytes + 255) & ~(size_t)255;
        return p;
    };
    float*     deg  = (float*)alloc((size_t)N * 4);
    float*     dis  = (float*)alloc((size_t)N * 4);
    float*     dinv = (float*)alloc((size_t)N * 4);
    _Float16*  W1t  = (_Float16*)alloc((size_t)D * D * 2);
    _Float16*  W2t  = (_Float16*)alloc((size_t)D * D * 2);
    _Float16*  Wft  = (_Float16*)alloc((size_t)D * D * 2);
    _Float16*  xh   = (_Float16*)alloc((size_t)N * D * 2);   // f16 activations (x, then relu(h1))
    float*     h    = (float*)alloc((size_t)N * D * 4);      // GEMM output (per conv)
    float*     agg  = (float*)alloc((size_t)N * D * 4);      // scatter accumulator
    float*     res  = (float*)alloc((size_t)N * D * 4);      // x @ Wfc

    const long long ND = (long long)N * D;
    const int nRowTiles   = (N + 15) / 16;                   // 3125 (N divisible by 16)
    const int gemmBlocks  = (nRowTiles + 7) / 8;             // 8 waves per 256-thread block
    const int scatBlocks  = gb((long long)E * 32);

    // degree (with self loop) and normalizers
    fill_f32<<<gb(N), 256, 0, stream>>>(deg, 1.0f, N);
    deg_accum<<<gb(E), 256, 0, stream>>>(deg, dstI, E);
    deg_finalize<<<gb(N), 256, 0, stream>>>(deg, dis, dinv, N);

    // weight transposes + f16 conversion
    convert_w_t<<<64, 256, 0, stream>>>(W1, W1t);
    convert_w_t<<<64, 256, 0, stream>>>(W2, W2t);
    convert_w_t<<<64, 256, 0, stream>>>(Wfc, Wft);
    convert_f32_to_f16<<<gb(ND), 256, 0, stream>>>(x, xh, (int)ND);

    // res = x @ Wfc  (bfc added in final epilogue)
    gemm_wmma<<<gemmBlocks, 256, 0, stream>>>(xh, Wft, res, nRowTiles);

    // ---- conv 1 ----
    gemm_wmma<<<gemmBlocks, 256, 0, stream>>>(xh, W1t, h, nRowTiles);
    fill_f32<<<gb(ND), 256, 0, stream>>>(agg, 0.0f, (int)ND);
    scatter_edges<<<scatBlocks, 256, 0, stream>>>(h, srcI, dstI, dis, agg, E);
    combine_relu_f16<<<gb(ND), 256, 0, stream>>>(agg, h, dinv, b1, xh, (int)ND);

    // ---- conv 2 ----
    gemm_wmma<<<gemmBlocks, 256, 0, stream>>>(xh, W2t, h, nRowTiles);
    fill_f32<<<gb(ND), 256, 0, stream>>>(agg, 0.0f, (int)ND);
    scatter_edges<<<scatBlocks, 256, 0, stream>>>(h, srcI, dstI, dis, agg, E);

    // out = relu(conv2 + res)
    combine_final<<<gb(ND), 256, 0, stream>>>(agg, h, dinv, b2, res, bfc, out, (int)ND);
}
